// ReverbNetwork_50371376448004
// MI455X (gfx1250) — matrix operand order repaired and checked by reading, not compiled, not run on hardware
//
#include <hip/hip_runtime.h>
#include <hip/hip_fp16.h>

typedef __attribute__((ext_vector_type(16))) _Float16 v16h;
typedef __attribute__((ext_vector_type(4)))  _Float16 v4h;
typedef __attribute__((ext_vector_type(8)))  float    v8f;

#define NN   24            // real nodes
#define NB   4
#define NC   8
#define NH   96
#define NW   96
#define NE   96
#define HW   (NH*NW)       // 9216
#define BCHW (NB*NC*HW)    // 294912
#define WKK  (NC*NC*16)    // per-edge weight elements = 1024
// padded activation planes: yp = y+1 in [0,98], xp = x+1 in [0,98] -> 100x100
#define HP   100
#define WP   100
#define PLANE (HP*WP)      // 10000
#define TILES_PER_B (NH*(NW/16))     // 576
#define NTILES      (NB*TILES_PER_B) // 2304
#define NSLOTS 288                   // 72 blocks * 4 waves
#define TPW    8                     // NTILES / NSLOTS

// ---------- prep: padded act_h = (half) sigmoid(states), zero halo ----------
__global__ void prep_act_kernel(const float* __restrict__ states,
                                _Float16* __restrict__ act, int n) {
    int i = blockIdx.x * blockDim.x + threadIdx.x;
    if (i < n) {
        int plane = i / PLANE;
        int rem   = i - plane * PLANE;
        int yp    = rem / WP;
        int xp    = rem - yp * WP;
        _Float16 v = (_Float16)0.f;
        if (yp >= 1 && yp <= NH && xp >= 1 && xp <= NW) {
            float s = states[(size_t)plane * HW + (yp - 1) * NW + (xp - 1)];
            v = (_Float16)(1.0f / (1.0f + __expf(-s)));
        }
        act[i] = v;
    }
}

// ---------- prep: w_h = (half)(weights * plasticity[e]) ----------
__global__ void prep_w_kernel(const float* __restrict__ w,
                              const float* __restrict__ plast,
                              _Float16* __restrict__ wh) {
    int i = blockIdx.x * blockDim.x + threadIdx.x;
    if (i < NE * WKK) {
        int e = i / WKK;
        wh[i] = (_Float16)(w[i] * plast[e]);
    }
}

// ---------- main conv via WMMA implicit GEMM ----------
// grid = (72, NN); 288 waves per node, each owns TPW=8 16x8 output tiles.
// Edge loop outer: B fragments (contiguous v16h) built once per edge;
// A fragments are 4x 8-byte unconditional loads from the padded planes.
__global__ void __launch_bounds__(128)
conv_wmma_kernel(const _Float16* __restrict__ act,
                 const _Float16* __restrict__ wh,
                 const int* __restrict__ src,
                 const int* __restrict__ dst,
                 float* __restrict__ out) {
    __shared__ int s_e[NE], s_s[NE];
    __shared__ int s_deg;
    const int node = blockIdx.y;
    if (threadIdx.x == 0) {                 // serial build => deterministic order
        int d = 0;
        for (int t = 0; t < NE; ++t)
            if (dst[t] == node) { s_e[d] = t; s_s[d] = src[t]; ++d; }
        s_deg = d;
    }
    __syncthreads();
    const int   deg    = s_deg;
    const float invdeg = 1.0f / (float)(deg > 0 ? deg : 1);

    const int lane = threadIdx.x & 31;
    const int wid  = threadIdx.x >> 5;
    const int m    = lane & 15;      // A row (spatial) / B column (cout)
    const int hi   = lane >> 4;      // half-wave select
    const int slot = blockIdx.x * 4 + wid;   // 0..287

    int tb[TPW], ty[TPW], tx[TPW];
    #pragma unroll
    for (int s = 0; s < TPW; ++s) {
        int t = slot + s * NSLOTS;
        int b = t / TILES_PER_B;
        int r = t - b * TILES_PER_B;
        int y = r / 6;
        tb[s] = b; ty[s] = y; tx[s] = (r - y * 6) * 16;
    }

    v8f acc[TPW];
    #pragma unroll
    for (int s = 0; s < TPW; ++s) acc[s] = (v8f){};

    for (int ei = 0; ei < deg; ++ei) {
        const int e  = s_e[ei];
        const int sn = s_s[ei];

        // B fragments: lane m holds K 0..15 (hi=0) / 16..31 (hi=1) of chunk c,
        // i.e. one contiguous, 32B-aligned run of 16 halves.
        const _Float16* __restrict__ wb = wh + (size_t)e * WKK;
        v16h vb[4];
        #pragma unroll
        for (int c = 0; c < 4; ++c) {
            v16h tmp = {};
            if (m < NC)
                tmp = *(const v16h*)(wb + (m * NC + c * 2 + hi) * 16);
            vb[c] = tmp;
        }

        const _Float16* __restrict__ an = act + (size_t)sn * (NB * NC * PLANE);
        #pragma unroll
        for (int s = 0; s < TPW; ++s) {
            const _Float16* __restrict__ abase =
                an + (size_t)tb[s] * (NC * PLANE);
            const int xcol = tx[s] + m;          // padded col of kx=0 tap
            const int yrow = ty[s];              // padded row of first tap
            #pragma unroll
            for (int c = 0; c < 4; ++c) {
                const _Float16* __restrict__ p0 =
                    abase + (c * 2 + hi) * PLANE + yrow * WP + xcol;
                v4h g0, g1, g2, g3;              // 4 contiguous kx taps each
                __builtin_memcpy(&g0, p0,              8);
                __builtin_memcpy(&g1, p0 + WP,         8);
                __builtin_memcpy(&g2, p0 + 2 * WP,     8);
                __builtin_memcpy(&g3, p0 + 3 * WP,     8);
                v16h va;
                #pragma unroll
                for (int q = 0; q < 4; ++q) {
                    va[q]      = g0[q];
                    va[4 + q]  = g1[q];
                    va[8 + q]  = g2[q];
                    va[12 + q] = g3[q];
                }
                acc[s] = __builtin_amdgcn_wmma_f32_16x16x32_f16(
                           false, va, false, vb[c], (short)0, acc[s],
                           false, false);
            }
        }
    }

    // D layout: lane column N = m (cout), rows M = hi*8 + v (v = VGPR idx)
    #pragma unroll
    for (int s = 0; s < TPW; ++s) {
        if (m < NC) {
            float* __restrict__ ob =
                out + (((size_t)node * NB + tb[s]) * NC + m) * HW
                    + ty[s] * NW + tx[s] + hi * 8;
            float4 o0, o1;
            o0.x = acc[s][0] * invdeg; o0.y = acc[s][1] * invdeg;
            o0.z = acc[s][2] * invdeg; o0.w = acc[s][3] * invdeg;
            o1.x = acc[s][4] * invdeg; o1.y = acc[s][5] * invdeg;
            o1.z = acc[s][6] * invdeg; o1.w = acc[s][7] * invdeg;
            *(float4*)(ob)     = o0;
            *(float4*)(ob + 4) = o1;
        }
    }
}

// ---------- per-node mean of sigmoid(future): deterministic tree reduce ----------
__global__ void node_mean_kernel(const float* __restrict__ fut,
                                 float* __restrict__ mean) {
    __shared__ float sdata[256];
    const int n = blockIdx.x;
    const float* __restrict__ p = fut + (size_t)n * BCHW;
    float s = 0.f;
    for (int i = threadIdx.x; i < BCHW; i += 256) {
        float v = p[i];
        s += 1.0f / (1.0f + __expf(-v));
    }
    sdata[threadIdx.x] = s;
    __syncthreads();
    for (int off = 128; off > 0; off >>= 1) {
        if (threadIdx.x < off) sdata[threadIdx.x] += sdata[threadIdx.x + off];
        __syncthreads();
    }
    if (threadIdx.x == 0) mean[n] = sdata[0] * (1.0f / (float)BCHW);
}

// ---------- new_plasticity ----------
__global__ void plast_kernel(const float* __restrict__ plast,
                             const int* __restrict__ dst,
                             const float* __restrict__ mean,
                             float* __restrict__ outp) {
    int e = blockIdx.x * blockDim.x + threadIdx.x;
    if (e < NE) outp[e] = plast[e] + 0.1f * (mean[dst[e]] - 0.5f);
}

extern "C" void kernel_launch(void* const* d_in, const int* in_sizes, int n_in,
                              void* d_out, int out_size, void* d_ws, size_t ws_size,
                              hipStream_t stream) {
    const float* states  = (const float*)d_in[0];
    const float* weights = (const float*)d_in[1];
    const float* plast   = (const float*)d_in[2];
    const int*   src     = (const int*)d_in[3];
    const int*   dst     = (const int*)d_in[4];
    float*       out     = (float*)d_out;

    char* ws = (char*)d_ws;
    _Float16* act_p = (_Float16*)ws;                       // 25*32*PLANE halves = 16,000,000 B
    _Float16* w_h   = (_Float16*)(ws + 16000000);          // NE*WKK halves      =    196,608 B
    float*    mean  = (float*)(ws + 16000000 + 196608);    // 24 floats

    const int nact = (NN + 1) * NB * NC * PLANE;           // 8,000,000
    prep_act_kernel<<<(nact + 255) / 256, 256, 0, stream>>>(states, act_p, nact);
    prep_w_kernel<<<(NE * WKK + 255) / 256, 256, 0, stream>>>(weights, plast, w_h);
    conv_wmma_kernel<<<dim3(72, NN), 128, 0, stream>>>(act_p, w_h, src, dst, out);
    node_mean_kernel<<<NN, 256, 0, stream>>>(out, mean);
    plast_kernel<<<1, 128, 0, stream>>>(plast, dst, mean, out + (size_t)NN * BCHW);
}
